// ForgettingAttention_65240553226416
// MI455X (gfx1250) — matrix operand order repaired and checked by reading, not compiled, not run on hardware
//
#include <hip/hip_runtime.h>
#include <hip/hip_bf16.h>

// ---------------------------------------------------------------------------
// Forgetting attention forward for MI455X (gfx1250, wave32, WMMA).
// B=2, N=2048, D=1024, H=16, HD=64.
// ---------------------------------------------------------------------------

#define BB 2
#define NN 2048
#define DD 1024
#define HH 16
#define HD 64
#define NEGBIG (-1e30f)

typedef __attribute__((ext_vector_type(16))) __bf16 v16bf;
typedef __attribute__((ext_vector_type(8)))  float  v8f;

union FragBF { v16bf v; unsigned int u[8]; };

// ---------------------------------------------------------------------------
// CDNA5 async global->LDS copy (16B per call). The builtin's params are
// v4i32 pointers in addrspace(1) (global) and addrspace(3) (LDS).
// Guarded so compilation never breaks if the builtin is absent.
// ---------------------------------------------------------------------------
typedef int av4i __attribute__((vector_size(16)));

#if defined(__HIP_DEVICE_COMPILE__) && \
    __has_builtin(__builtin_amdgcn_global_load_async_to_lds_b128)
#define ASYNC_CP16(dst, src)                                            \
    __builtin_amdgcn_global_load_async_to_lds_b128(                     \
        (__attribute__((address_space(1))) av4i*)(src),                 \
        (__attribute__((address_space(3))) av4i*)(dst), 0, 0)
#if __has_builtin(__builtin_amdgcn_s_wait_asynccnt)
#define ASYNC_WAIT() __builtin_amdgcn_s_wait_asynccnt(0)
#else
#define ASYNC_WAIT() asm volatile("s_wait_asynccnt 0x0" ::: "memory")
#endif
#else
#define ASYNC_CP16(dst, src) (*(uint4*)(dst) = *(const uint4*)(src))
#define ASYNC_WAIT() ((void)0)
#endif

// A operand, this code's contiguous k order (not AMD's table order): lane m = l&15, half = l>>4
// VGPR v holds packed K pair at kbase = half*16 + v*2
__device__ __forceinline__ int a_kbase(int v, int half) {
    return (half << 4) + (v << 1);
}
// B operand, this code's contiguous k order (not AMD's table order): lane n = l&15, half = l>>4
// VGPR v holds packed K pair at kbase = half*16 + v*2
__device__ __forceinline__ int b_kbase(int v, int half) {
    return (half << 4) + (v << 1);
}

__device__ __forceinline__ unsigned short f32_to_bf16(float f) {
    unsigned int u = __float_as_uint(f);
    unsigned int lsb = (u >> 16) & 1u;
    u += 0x7fffu + lsb;           // round to nearest even
    return (unsigned short)(u >> 16);
}

__device__ __forceinline__ unsigned int pack_bf16(float lo, float hi) {
    return (unsigned int)f32_to_bf16(lo) | ((unsigned int)f32_to_bf16(hi) << 16);
}

// ---------------------------------------------------------------------------
// Kernel: f32 -> bf16 straight convert
// ---------------------------------------------------------------------------
__global__ void k_convert(const float* __restrict__ in,
                          unsigned short* __restrict__ out, int n) {
    int i = blockIdx.x * blockDim.x + threadIdx.x;
    if (i < n) out[i] = f32_to_bf16(in[i]);
}

// ---------------------------------------------------------------------------
// Kernel: transpose [K x N] f32 -> [N x K] bf16  (so B-fragments load as
// contiguous uint pairs along K)
// ---------------------------------------------------------------------------
__global__ void k_transpose(const float* __restrict__ in,
                            unsigned short* __restrict__ out, int K, int N) {
    int i = blockIdx.x * blockDim.x + threadIdx.x;
    if (i >= K * N) return;
    int n = i / K, k = i % K;
    out[(size_t)n * K + k] = f32_to_bf16(in[(size_t)k * N + n]);
}

// ---------------------------------------------------------------------------
// Kernel: bf16 GEMM  C[M x N] = A[M x K] * Bt[N x K]^T
// Block = 8 waves = 128x64 macro tile. B tile (64x32) staged in LDS per
// k-step via async global->LDS and shared across all waves; each wave keeps
// a 16x64 f32 accumulator (4 WMMAs per k-step, A fragment reused 4x).
// ---------------------------------------------------------------------------
template <bool F32OUT>
__global__ void k_gemm_bf16(const unsigned short* __restrict__ A,
                            const unsigned short* __restrict__ Bt,
                            void* __restrict__ C, int M, int N, int K) {
    __shared__ unsigned short blds[64 * 32];     // Bt tile: 64 cols x 32 k

    const int lane = threadIdx.x & 31;
    const int wid  = threadIdx.x >> 5;
    const int half = lane >> 4;
    const int nlo  = lane & 15;

    const int nblk = N >> 6;                     // 64-col macro tiles
    const int bm   = blockIdx.x / nblk;          // 128-row macro tile
    const int bn   = blockIdx.x % nblk;

    const size_t arow = (size_t)(bm * 128 + wid * 16 + nlo) * K;

    // staging assignment: thread t copies 16B: Bt row sr, k-offset sc
    const int tid = threadIdx.x;
    const int sr  = tid >> 2;                    // 0..63
    const int sc  = (tid & 3) << 3;              // 0,8,16,24 (elements)
    const size_t brow = (size_t)(bn * 64 + sr) * K;

    v8f z = {0.f, 0.f, 0.f, 0.f, 0.f, 0.f, 0.f, 0.f};
    v8f acc[4] = {z, z, z, z};

    for (int kk = 0; kk < K; kk += 32) {
        // stage B tile (4 KB) into LDS
        ASYNC_CP16(blds + sr * 32 + sc, Bt + brow + kk + sc);
        ASYNC_WAIT();
        __syncthreads();

        if (kk + 32 < K)
            __builtin_prefetch(A + arow + kk + 32, 0, 0);

        FragBF a;
#pragma unroll
        for (int v = 0; v < 8; ++v)
            a.u[v] = *(const unsigned int*)(A + arow + kk + a_kbase(v, half));

#pragma unroll
        for (int g = 0; g < 4; ++g) {
            FragBF b;
#pragma unroll
            for (int v = 0; v < 8; ++v)
                b.u[v] = *(const unsigned int*)(blds + (g * 16 + nlo) * 32 +
                                                b_kbase(v, half));
            acc[g] = __builtin_amdgcn_wmma_f32_16x16x32_bf16(
                false, a.v, false, b.v, (short)0, acc[g], false, false);
        }
        __syncthreads();
    }

#pragma unroll
    for (int g = 0; g < 4; ++g) {
        const int col = bn * 64 + g * 16 + nlo;
#pragma unroll
        for (int v = 0; v < 8; ++v) {
            const int row = bm * 128 + wid * 16 + half * 8 + v;
            if (F32OUT)
                ((float*)C)[(size_t)row * N + col] = acc[g][v];
            else
                ((unsigned short*)C)[(size_t)row * N + col] = f32_to_bf16(acc[g][v]);
        }
    }
}

// ---------------------------------------------------------------------------
// Kernel: c[b,h,n] = cumsum_n( log_sigmoid( x[b,n,:] @ Wf[:,h] ) )
// one block per (b,h); 256 threads do the dots, wave 0 does the scan.
// ---------------------------------------------------------------------------
__global__ void k_logf_cumsum(const float* __restrict__ x,
                              const float* __restrict__ Wf,
                              float* __restrict__ c) {
    __shared__ float lf[NN];
    const int b = blockIdx.x / HH;
    const int h = blockIdx.x % HH;
    const int tid = threadIdx.x;

    for (int n = tid; n < NN; n += blockDim.x) {
        const float* xr = x + ((size_t)b * NN + n) * DD;
        float dot = 0.f;
        for (int k = 0; k < DD; ++k) dot += xr[k] * Wf[(size_t)k * HH + h];
        // stable log-sigmoid
        float t = fabsf(dot);
        lf[n] = fminf(dot, 0.f) - log1pf(__expf(-t));
    }
    __syncthreads();

    if (tid < 32) {
        const int lane = tid;
        float running = 0.f;
        float* cout = c + ((size_t)b * HH + h) * NN;
        for (int base = 0; base < NN; base += 32) {
            float val = lf[base + lane];
#pragma unroll
            for (int off = 1; off < 32; off <<= 1) {
                float other = __shfl_up(val, off, 32);
                if (lane >= off) val += other;
            }
            cout[base + lane] = running + val;
            running += __shfl(val, 31, 32);
        }
    }
}

// ---------------------------------------------------------------------------
// Kernel: flash-style forgetting attention.
// One wave per (b, h, 16-row query tile). Streams key blocks of 32.
// QK^T : 4 WMMAs / block ; P@V : 4 WMMAs / block.
// ---------------------------------------------------------------------------
__global__ void k_fox_attn(const unsigned short* __restrict__ qb,
                           const unsigned short* __restrict__ kb,
                           const unsigned short* __restrict__ vb,
                           const float* __restrict__ cbuf,
                           unsigned short* __restrict__ attnb) {
    __shared__ unsigned short vlds[32 * HD];   // V tile  (32 keys x 64)
    __shared__ float          plds[16 * 32];   // P tile  (16 rows x 32 keys)

    const int itiles = NN / 16;
    const int it = blockIdx.x % itiles;
    const int h  = (blockIdx.x / itiles) % HH;
    const int b  = blockIdx.x / (itiles * HH);

    const int lane = threadIdx.x & 31;
    const int half = lane >> 4;
    const int nlo  = lane & 15;
    const float scale = 0.125f;   // HD^-0.5 = 1/8

    const float* crow = cbuf + ((size_t)b * HH + h) * NN;

    // preload Q fragments (head-dim 0..31 and 32..63)
    FragBF a0, a1;
    {
        const size_t qoff = ((size_t)(b * NN + it * 16 + nlo)) * DD + h * HD;
#pragma unroll
        for (int v = 0; v < 8; ++v) {
            a0.u[v] = *(const unsigned int*)(qb + qoff + a_kbase(v, half));
            a1.u[v] = *(const unsigned int*)(qb + qoff + 32 + a_kbase(v, half));
        }
    }

    // per-lane row state: rows r[v] = it*16 + half*8 + v
    float rM[8], rL[8], cq[8];
    int   rIdx[8];
#pragma unroll
    for (int v = 0; v < 8; ++v) {
        rIdx[v] = it * 16 + half * 8 + v;
        cq[v]   = crow[rIdx[v]];
        rM[v]   = -3.0e38f;
        rL[v]   = 0.f;
    }
    v8f accz = {0.f, 0.f, 0.f, 0.f, 0.f, 0.f, 0.f, 0.f};
    v8f acc[4] = {accz, accz, accz, accz};

    const int qmax = it * 16 + 15;
    for (int j32 = 0; j32 <= qmax; j32 += 32) {
        // ---- async-stage V (32 keys x head dims 0..31 only, 4x16B per key) into LDS early ----
        {
            const unsigned short* src =
                vb + ((size_t)(b * NN + j32 + lane)) * DD + h * HD;
            unsigned short* dst = vlds + lane * HD;
#pragma unroll
            for (int q4 = 0; q4 < 4; ++q4)
                ASYNC_CP16(dst + q4 * 8, src + q4 * 8);
        }

        // ---- scores: two 16x16 tiles (keys j32..+15 and j32+16..+31) ----
        v8f s[2];
#pragma unroll
        for (int sub = 0; sub < 2; ++sub) {
            FragBF b0, b1;
            const size_t koff =
                ((size_t)(b * NN + j32 + sub * 16 + nlo)) * DD + h * HD;
#pragma unroll
            for (int v = 0; v < 8; ++v) {
                b0.u[v] = *(const unsigned int*)(kb + koff + b_kbase(v, half));
                b1.u[v] = *(const unsigned int*)(kb + koff + 32 + b_kbase(v, half));
            }
            v8f t = {0.f, 0.f, 0.f, 0.f, 0.f, 0.f, 0.f, 0.f};
            t = __builtin_amdgcn_wmma_f32_16x16x32_bf16(
                false, a0.v, false, b0.v, (short)0, t, false, false);
            t = __builtin_amdgcn_wmma_f32_16x16x32_bf16(
                false, a1.v, false, b1.v, (short)0, t, false, false);
            s[sub] = t;
        }

        // ---- bias + causal mask ----
        const float cj0 = crow[j32 + nlo];
        const float cj1 = crow[j32 + 16 + nlo];
#pragma unroll
        for (int v = 0; v < 8; ++v) {
            float v0 = s[0][v] * scale + cq[v] - cj0;
            float v1 = s[1][v] * scale + cq[v] - cj1;
            if (j32 + nlo > rIdx[v])      v0 = NEGBIG;
            if (j32 + 16 + nlo > rIdx[v]) v1 = NEGBIG;
            s[0][v] = v0;
            s[1][v] = v1;
        }

        // ---- online softmax (row reductions across each 16-lane half) ----
        float alpha[8];
#pragma unroll
        for (int v = 0; v < 8; ++v) {
            float mx = fmaxf(s[0][v], s[1][v]);
#pragma unroll
            for (int off = 1; off < 16; off <<= 1)
                mx = fmaxf(mx, __shfl_xor(mx, off, 32));
            float newM = fmaxf(rM[v], mx);
            alpha[v] = __expf(rM[v] - newM);
            float p0 = __expf(s[0][v] - newM);
            float p1 = __expf(s[1][v] - newM);
            s[0][v] = p0;
            s[1][v] = p1;
            float rs = p0 + p1;
#pragma unroll
            for (int off = 1; off < 16; off <<= 1)
                rs += __shfl_xor(rs, off, 32);
            rL[v] = rL[v] * alpha[v] + rs;
            rM[v] = newM;
        }
#pragma unroll
        for (int g = 0; g < 4; ++g)
#pragma unroll
            for (int v = 0; v < 8; ++v) acc[g][v] *= alpha[v];

        // ---- relayout P (C-layout f32) -> A-fragment bf16 via LDS ----
#pragma unroll
        for (int v = 0; v < 8; ++v) {
            plds[(half * 8 + v) * 32 + nlo]      = s[0][v];
            plds[(half * 8 + v) * 32 + 16 + nlo] = s[1][v];
        }
        ASYNC_WAIT();
        __syncthreads();

        FragBF pf;
#pragma unroll
        for (int v = 0; v < 8; ++v) {
            const int kbp = a_kbase(v, half);
            pf.u[v] = pack_bf16(plds[nlo * 32 + kbp], plds[nlo * 32 + kbp + 1]);
        }

        // ---- O += P @ V  (4 groups of 16 output cols) ----
#pragma unroll
        for (int g = 0; g < 4; ++g) {
            FragBF vf;
#pragma unroll
            for (int v = 0; v < 8; ++v) {
                const int kbv = b_kbase(v, half);
                vf.u[v] = (unsigned int)vlds[kbv * HD + g * 16 + nlo] |
                          ((unsigned int)vlds[(kbv + 1) * HD + g * 16 + nlo] << 16);
            }
            acc[g] = __builtin_amdgcn_wmma_f32_16x16x32_bf16(
                false, pf.v, false, vf.v, (short)0, acc[g], false, false);
        }
        __syncthreads();
    }

    // ---- normalize and write merged-head bf16 output [b, n, D] ----
    float inv[8];
#pragma unroll
    for (int v = 0; v < 8; ++v) inv[v] = 1.f / rL[v];
#pragma unroll
    for (int g = 0; g < 4; ++g) {
#pragma unroll
        for (int v = 0; v < 8; ++v) {
            const int row = rIdx[v];
            const int col = h * HD + g * 16 + nlo;
            attnb[((size_t)(b * NN + row)) * DD + col] =
                f32_to_bf16(acc[g][v] * inv[v]);
        }
    }
}

// ---------------------------------------------------------------------------
// Host launcher
// ---------------------------------------------------------------------------
extern "C" void kernel_launch(void* const* d_in, const int* in_sizes, int n_in,
                              void* d_out, int out_size, void* d_ws, size_t ws_size,
                              hipStream_t stream) {
    (void)in_sizes; (void)n_in; (void)out_size; (void)ws_size;

    const float* x  = (const float*)d_in[0];
    const float* Wq = (const float*)d_in[1];
    const float* Wk = (const float*)d_in[2];
    const float* Wv = (const float*)d_in[3];
    const float* Wo = (const float*)d_in[4];
    const float* Wf = (const float*)d_in[5];
    float* out = (float*)d_out;

    // workspace layout (elements)
    unsigned short* w = (unsigned short*)d_ws;
    const size_t XE = (size_t)BB * NN * DD;      // 4,194,304
    const size_t WE = (size_t)DD * DD;           // 1,048,576
    unsigned short* xbf   = w;
    unsigned short* wqT   = xbf + XE;
    unsigned short* wkT   = wqT + WE;
    unsigned short* wvT   = wkT + WE;
    unsigned short* woT   = wvT + WE;
    unsigned short* qb    = woT + WE;
    unsigned short* kbuf  = qb + XE;
    unsigned short* vbuf  = kbuf + XE;
    unsigned short* attnb = vbuf + XE;
    float* cbuf = (float*)(attnb + XE);          // B*H*N floats

    const int M = BB * NN;   // 4096

    // 1. converts / transposes
    k_convert<<<(int)((XE + 255) / 256), 256, 0, stream>>>(x, xbf, (int)XE);
    k_transpose<<<(int)((WE + 255) / 256), 256, 0, stream>>>(Wq, wqT, DD, DD);
    k_transpose<<<(int)((WE + 255) / 256), 256, 0, stream>>>(Wk, wkT, DD, DD);
    k_transpose<<<(int)((WE + 255) / 256), 256, 0, stream>>>(Wv, wvT, DD, DD);
    k_transpose<<<(int)((WE + 255) / 256), 256, 0, stream>>>(Wo, woT, DD, DD);

    // 2. Q/K/V projections (bf16 WMMA GEMM, bf16 out)
    const int gblocks = (M / 128) * (DD / 64);   // 512 blocks, 8 waves each
    k_gemm_bf16<false><<<gblocks, 256, 0, stream>>>(xbf, wqT, qb,   M, DD, DD);
    k_gemm_bf16<false><<<gblocks, 256, 0, stream>>>(xbf, wkT, kbuf, M, DD, DD);
    k_gemm_bf16<false><<<gblocks, 256, 0, stream>>>(xbf, wvT, vbuf, M, DD, DD);

    // 3. forget-gate cumulative sums
    k_logf_cumsum<<<BB * HH, 256, 0, stream>>>(x, Wf, cbuf);

    // 4. flash forgetting-attention (one wave per 16-row query tile)
    k_fox_attn<<<BB * HH * (NN / 16), 32, 0, stream>>>(qb, kbuf, vbuf, cbuf, attnb);

    // 5. output projection (f32 out)
    k_gemm_bf16<true><<<gblocks, 256, 0, stream>>>(attnb, woT, out, M, DD, DD);
}
